// ExpSelfAttention_317827580123
// MI455X (gfx1250) — compile-verified, matrix operand order checked by reading, not run on hardware
//
#include <hip/hip_runtime.h>
#include <hip/hip_bf16.h>

// ---------------------------------------------------------------------------
// ExpSelfAttention block for MI455X (gfx1250), wave32 + WMMA bf16.
//  x:[2048,16,512] f32. attn uses exp-decay causal matrix -> linear scan.
//  GEMMs: software-pipelined (double-buffered register fragments) so WMMAs
//  issue back-to-back instead of stalling on s_wait_loadcnt each step.
// ---------------------------------------------------------------------------

#define SEQ     2048
#define BATCH   16
#define DMODEL  512
#define FFDIM   2048
#define MROWS   (SEQ * BATCH)          // 32768
#define COLS    (BATCH * DMODEL)       // 8192 scan columns
#define ALPHA   0.9f
#define LNEPS   1e-5f
#define CHUNKS  32
#define CLEN    (SEQ / CHUNKS)         // 64

typedef __bf16 bf16;
typedef __attribute__((ext_vector_type(8)))  __bf16 bf16x8;
typedef __attribute__((ext_vector_type(16))) __bf16 bf16x16;
typedef __attribute__((ext_vector_type(8)))  float  v8f;
typedef __attribute__((ext_vector_type(4)))  float  f32x4;

// ---------------- fp32 -> bf16 weight conversion ---------------------------
__global__ __launch_bounds__(256) void cvt_bf16_kernel(const float* __restrict__ a,
                                                       bf16* __restrict__ o, int n) {
    int i = blockIdx.x * 256 + threadIdx.x;
    if (i < n) o[i] = (bf16)a[i];
}

// ---------------- LayerNorm (D=512) -> bf16, one wave32 per row ------------
__global__ __launch_bounds__(256) void ln_bf16_kernel(const float* __restrict__ x,
                                                      const float* __restrict__ g,
                                                      const float* __restrict__ be,
                                                      bf16* __restrict__ y, int rows) {
    int row  = blockIdx.x * 8 + (threadIdx.x >> 5);
    int lane = threadIdx.x & 31;
    if (row >= rows) return;
    const f32x4* xp = (const f32x4*)(x + (size_t)row * DMODEL + lane * 16);
    f32x4 q0 = xp[0], q1 = xp[1], q2 = xp[2], q3 = xp[3];
    float s = 0.f, s2 = 0.f;
    float vals[16];
#pragma unroll
    for (int j = 0; j < 4; ++j) {
        vals[j]      = q0[j];
        vals[4 + j]  = q1[j];
        vals[8 + j]  = q2[j];
        vals[12 + j] = q3[j];
    }
#pragma unroll
    for (int j = 0; j < 16; ++j) { s += vals[j]; s2 += vals[j] * vals[j]; }
#pragma unroll
    for (int o = 16; o > 0; o >>= 1) {
        s  += __shfl_xor(s, o);
        s2 += __shfl_xor(s2, o);
    }
    float mean = s * (1.f / DMODEL);
    float var  = s2 * (1.f / DMODEL) - mean * mean;
    float rs   = rsqrtf(var + LNEPS);
    int c0 = lane * 16;
    bf16x8 o0, o1;
#pragma unroll
    for (int j = 0; j < 8; ++j) {
        o0[j] = (bf16)((vals[j]     - mean) * rs * g[c0 + j]     + be[c0 + j]);
        o1[j] = (bf16)((vals[8 + j] - mean) * rs * g[c0 + 8 + j] + be[c0 + 8 + j]);
    }
    bf16x8* yp = (bf16x8*)(y + (size_t)row * DMODEL + c0);
    yp[0] = o0;
    yp[1] = o1;
}

// ---------------- WMMA bf16 GEMM:  C[M,N] = A[M,K] * Bw[N,K]^T (+bias) -----
__device__ inline bf16x16 load_frag(const bf16* p) {
    // ISA 16-bit A/B layout: per lane two contiguous 8-element chunks, 32B apart
    bf16x8 lo = *(const bf16x8*)p;
    bf16x8 hi = *(const bf16x8*)(p + 16);
    return __builtin_shufflevector(lo, hi, 0, 1, 2, 3, 4, 5, 6, 7,
                                   8, 9, 10, 11, 12, 13, 14, 15);
}

__device__ inline v8f wmma_bf16(bf16x16 a, bf16x16 b, v8f c) {
    return __builtin_amdgcn_wmma_f32_16x16x32_bf16(
        false, a, false, b, (short)0, c, false, false);
}

// Wave tile: 32(M) x 64(N) = 8 accumulators; 8 waves per block as 4Mx2N
// -> 128x128 block tile. K-loop double-buffers all 6 fragments in registers.
template <bool RELU, bool RESID, bool OUT_BF16>
__global__ __launch_bounds__(256) void gemm_wmma_kernel(
    const bf16* __restrict__ A,      // [M,K] row-major bf16 (activations)
    const bf16* __restrict__ Bw,     // [N,K] row-major bf16 (weights, out-major)
    const float* __restrict__ bias,  // [N]
    const float* __restrict__ resid, // [M,N] or null
    float* __restrict__ Cf, bf16* __restrict__ Cb,
    int M, int N, int K) {
    const int lane  = threadIdx.x & 31;
    const int wave  = threadIdx.x >> 5;
    const int waveM = wave >> 1;            // 0..3
    const int waveN = wave & 1;             // 0..1
    const int tileM = blockIdx.y * 128 + waveM * 32;
    const int tileN = blockIdx.x * 128 + waveN * 64;
    const int r15   = lane & 15;
    const int h     = lane >> 4;

    const bf16* ap0 = A + (size_t)(tileM + r15) * K + h * 8;
    const bf16* ap1 = ap0 + (size_t)16 * K;
    const bf16* bp0 = Bw + (size_t)(tileN + r15) * K + h * 8;
    const bf16* bp1 = bp0 + (size_t)16 * K;
    const bf16* bp2 = bp0 + (size_t)32 * K;
    const bf16* bp3 = bp0 + (size_t)48 * K;

    v8f acc[8] = {v8f{}, v8f{}, v8f{}, v8f{}, v8f{}, v8f{}, v8f{}, v8f{}};

    // prologue: load first K-step fragments
    bf16x16 a0c = load_frag(ap0);
    bf16x16 a1c = load_frag(ap1);
    bf16x16 b0c = load_frag(bp0);
    bf16x16 b1c = load_frag(bp1);
    bf16x16 b2c = load_frag(bp2);
    bf16x16 b3c = load_frag(bp3);

#pragma unroll 2
    for (int kk = 32; kk < K; kk += 32) {
        // issue next-iteration loads first; WMMAs below consume registers
        // already resident, so waits for these loads sink past the WMMAs.
        bf16x16 a0n = load_frag(ap0 + kk);
        bf16x16 a1n = load_frag(ap1 + kk);
        bf16x16 b0n = load_frag(bp0 + kk);
        bf16x16 b1n = load_frag(bp1 + kk);
        bf16x16 b2n = load_frag(bp2 + kk);
        bf16x16 b3n = load_frag(bp3 + kk);

        acc[0] = wmma_bf16(a0c, b0c, acc[0]);
        acc[1] = wmma_bf16(a0c, b1c, acc[1]);
        acc[2] = wmma_bf16(a0c, b2c, acc[2]);
        acc[3] = wmma_bf16(a0c, b3c, acc[3]);
        acc[4] = wmma_bf16(a1c, b0c, acc[4]);
        acc[5] = wmma_bf16(a1c, b1c, acc[5]);
        acc[6] = wmma_bf16(a1c, b2c, acc[6]);
        acc[7] = wmma_bf16(a1c, b3c, acc[7]);

        a0c = a0n; a1c = a1n;
        b0c = b0n; b1c = b1n; b2c = b2n; b3c = b3n;
    }
    // epilogue K-step
    acc[0] = wmma_bf16(a0c, b0c, acc[0]);
    acc[1] = wmma_bf16(a0c, b1c, acc[1]);
    acc[2] = wmma_bf16(a0c, b2c, acc[2]);
    acc[3] = wmma_bf16(a0c, b3c, acc[3]);
    acc[4] = wmma_bf16(a1c, b0c, acc[4]);
    acc[5] = wmma_bf16(a1c, b1c, acc[5]);
    acc[6] = wmma_bf16(a1c, b2c, acc[6]);
    acc[7] = wmma_bf16(a1c, b3c, acc[7]);

#pragma unroll
    for (int mf = 0; mf < 2; ++mf) {
#pragma unroll
        for (int nf = 0; nf < 4; ++nf) {
            const v8f& a = acc[mf * 4 + nf];
            int col  = tileN + nf * 16 + r15;
            float bv = bias[col];
#pragma unroll
            for (int r = 0; r < 8; ++r) {
                int row = tileM + mf * 16 + h * 8 + r;  // C layout: VGPR r -> M=r / r+8
                float v = a[r] + bv;
                if (RELU) v = fmaxf(v, 0.f);
                size_t idx = (size_t)row * N + col;
                if (RESID) v += resid[idx];
                if (OUT_BF16) Cb[idx] = (bf16)v;
                else          Cf[idx] = v;
            }
        }
    }
}

// ---------------- exp-decay causal mixing as a chunked linear scan ---------
// S[i] = a*(S[i-1] + z[i]);  attn[i] = S[i] + (1 - sum_{k=1}^{i+1} a^k) * z[i]
__global__ __launch_bounds__(256) void scan_phaseA(const float* __restrict__ z,
                                                   float* __restrict__ carry) {
    int col   = blockIdx.x * 256 + threadIdx.x;
    int chunk = blockIdx.y;
    const float* zp = z + (size_t)(chunk * CLEN) * COLS + col;
    float Sv = 0.f;
#pragma unroll 4
    for (int s = 0; s < CLEN; ++s) Sv = ALPHA * (Sv + zp[(size_t)s * COLS]);
    carry[(size_t)chunk * COLS + col] = Sv;
}

__global__ __launch_bounds__(256) void scan_phaseB(float* __restrict__ carry) {
    int col = blockIdx.x * 256 + threadIdx.x;
    float f = powf(ALPHA, (float)CLEN);  // exact cross-chunk decay
    float Sv = 0.f;
    for (int c = 0; c < CHUNKS; ++c) {
        size_t i = (size_t)c * COLS + col;
        float e  = carry[i];
        carry[i] = Sv;                   // exclusive carry-in
        Sv = f * Sv + e;
    }
}

__global__ __launch_bounds__(256) void scan_phaseC(const float* __restrict__ x,
                                                   const float* __restrict__ z,
                                                   const float* __restrict__ carry,
                                                   float* __restrict__ out) {
    int col   = blockIdx.x * 256 + threadIdx.x;
    int chunk = blockIdx.y;
    int s0    = chunk * CLEN;
    float Sv  = carry[(size_t)chunk * COLS + col];
    float p   = powf(ALPHA, (float)(s0 + 1));             // a^{i+1} at i=s0
    float cum = ALPHA * (1.f - p) / (1.f - ALPHA);        // sum_{k=1}^{s0+1} a^k
#pragma unroll 4
    for (int s = 0; s < CLEN; ++s) {
        size_t idx = (size_t)(s0 + s) * COLS + col;
        float zv = z[idx];
        Sv = ALPHA * (Sv + zv);
        out[idx] = x[idx] + Sv + (1.f - cum) * zv;
        p *= ALPHA;
        cum += p;
    }
}

// ---------------------------------------------------------------------------
extern "C" void kernel_launch(void* const* d_in, const int* in_sizes, int n_in,
                              void* d_out, int out_size, void* d_ws, size_t ws_size,
                              hipStream_t stream) {
    const float* x     = (const float*)d_in[0];
    const float* w_lin = (const float*)d_in[1];
    const float* b_lin = (const float*)d_in[2];
    const float* w1    = (const float*)d_in[3];
    const float* b1    = (const float*)d_in[4];
    const float* w2    = (const float*)d_in[5];
    const float* b2    = (const float*)d_in[6];
    const float* g1    = (const float*)d_in[7];
    const float* be1   = (const float*)d_in[8];
    const float* g2    = (const float*)d_in[9];
    const float* be2   = (const float*)d_in[10];
    float* out = (float*)d_out;

    char* ws = (char*)d_ws;
    size_t off = 0;
    auto alloc = [&](size_t bytes) -> void* {
        void* p = ws + off;
        off = (off + bytes + 255) & ~(size_t)255;
        return p;
    };
    bf16*  wlinB = (bf16*)alloc((size_t)DMODEL * DMODEL * 2);
    bf16*  w1B   = (bf16*)alloc((size_t)FFDIM * DMODEL * 2);
    bf16*  w2B   = (bf16*)alloc((size_t)DMODEL * FFDIM * 2);
    bf16*  yb    = (bf16*)alloc((size_t)MROWS * DMODEL * 2);
    float* z     = (float*)alloc((size_t)MROWS * DMODEL * 4);
    bf16*  hb    = (bf16*)alloc((size_t)MROWS * FFDIM * 2);
    float* carry = (float*)alloc((size_t)CHUNKS * COLS * 4);
    (void)ws_size; (void)in_sizes; (void)n_in; (void)out_size;

    // 1) weights -> bf16 (tiny, L2-resident afterwards)
    cvt_bf16_kernel<<<(DMODEL * DMODEL) / 256, 256, 0, stream>>>(w_lin, wlinB, DMODEL * DMODEL);
    cvt_bf16_kernel<<<(FFDIM * DMODEL) / 256, 256, 0, stream>>>(w1, w1B, FFDIM * DMODEL);
    cvt_bf16_kernel<<<(DMODEL * FFDIM) / 256, 256, 0, stream>>>(w2, w2B, DMODEL * FFDIM);

    // 2) LN1 -> bf16 activations
    ln_bf16_kernel<<<MROWS / 8, 256, 0, stream>>>(x, g1, be1, yb, MROWS);

    // 3) z = yb * w_lin^T + b_lin   (WMMA bf16, f32 accumulate/out)
    gemm_wmma_kernel<false, false, false>
        <<<dim3(DMODEL / 128, MROWS / 128), 256, 0, stream>>>(
            yb, wlinB, b_lin, nullptr, z, nullptr, MROWS, DMODEL, DMODEL);

    // 4) x1 = x + attn(z)  via chunked linear scan (exact carry propagation)
    scan_phaseA<<<dim3(COLS / 256, CHUNKS), 256, 0, stream>>>(z, carry);
    scan_phaseB<<<COLS / 256, 256, 0, stream>>>(carry);
    scan_phaseC<<<dim3(COLS / 256, CHUNKS), 256, 0, stream>>>(x, z, carry, out);

    // 5) LN2 -> bf16
    ln_bf16_kernel<<<MROWS / 8, 256, 0, stream>>>(out, g2, be2, yb, MROWS);

    // 6) h = relu(yb * w1^T + b1)  -> bf16 (feeds next GEMM)
    gemm_wmma_kernel<true, false, true>
        <<<dim3(FFDIM / 128, MROWS / 128), 256, 0, stream>>>(
            yb, w1B, b1, nullptr, nullptr, hb, MROWS, FFDIM, DMODEL);

    // 7) out = x1 + h * w2^T + b2  (residual read+write on d_out, same thread)
    gemm_wmma_kernel<false, true, false>
        <<<dim3(DMODEL / 128, MROWS / 128), 256, 0, stream>>>(
            hb, w2B, b2, out, out, nullptr, MROWS, DMODEL, FFDIM);
}
